// iDP3Encoder_40656160424210
// MI455X (gfx1250) — matrix operand order repaired and checked by reading, not compiled, not run on hardware
//
#include <hip/hip_runtime.h>
#include <stdint.h>

// ---------------- problem constants ----------------
#define BFB   16                 // B*F
#define NPTS  8192               // points per batch
#define HD    256                // hidden channels
#define NLAY  4
#define TOTP  (BFB * NPTS)       // 131072 points total
#define OUTC  1024

// ---------------- types ----------------
typedef __attribute__((ext_vector_type(16))) __bf16         v16bf;
typedef __attribute__((ext_vector_type(8)))  float          v8f;
typedef __attribute__((ext_vector_type(8)))  unsigned short v8us;

union FragU { v16bf bf; v8us us[2]; };

// ---------------- scalar helpers ----------------
__device__ __forceinline__ unsigned short f2bf(float f) {
  unsigned u = __float_as_uint(f);
  u += 0x7fffu + ((u >> 16) & 1u);          // round-to-nearest-even
  return (unsigned short)(u >> 16);
}
// order-preserving float->uint encoding (for atomicMax over signed floats)
__device__ __forceinline__ unsigned encf(float f) {
  unsigned u = __float_as_uint(f);
  return (u & 0x80000000u) ? ~u : (u | 0x80000000u);
}
__device__ __forceinline__ float decf(unsigned e) {
  return (e & 0x80000000u) ? __uint_as_float(e ^ 0x80000000u)
                           : __uint_as_float(~e);
}

// ---------------- WMMA fragment loads ----------------
// A fragment (16x32 bf16, 16-bit layout): lane<16 holds row M=lane, K 0-7 & 16-23;
// lane>=16 holds row M=lane-16, K 8-15 & 24-31.  rowbase = &X[row0*HD], row-major.
__device__ __forceinline__ v16bf load_a_frag(const unsigned short* __restrict__ rowbase,
                                             int k0, int lane) {
  const int m = lane & 15, half = lane >> 4;
  const unsigned short* p = rowbase + (size_t)m * HD + k0 + half * 8;
  FragU f;
  f.us[0] = *(const v8us*)(p);
  f.us[1] = *(const v8us*)(p + 16);
  return f.bf;
}
// B fragment from pre-swizzled weights: frag (ntile,ktile) stored as 32 lanes x 16 bf16,
// contiguous -> one coalesced 32B load per lane.
__device__ __forceinline__ v16bf load_b_frag(const unsigned short* __restrict__ W,
                                             int ntile, int ktile, int lane) {
  const unsigned short* p = W + (((size_t)ntile * 8 + ktile) << 9) + (lane << 4);
  FragU f;
  f.us[0] = *(const v8us*)(p);
  f.us[1] = *(const v8us*)(p + 8);
  return f.bf;
}

#define WMMA_BF16(a, b, c)                                                   \
  __builtin_amdgcn_wmma_f32_16x16x32_bf16(false, (a), false, (b), (short)0,  \
                                          (c), false, false)

// ---------------- weight pre-swizzle + workspace init ----------------
// Converts Wl / Wg[:, :H] / Wo(layer slices) into bf16 B-fragment order, and
// zeroes the per-layer channel-max and global-max buffers.
__global__ __launch_bounds__(256) void k_init(
    const float* __restrict__ Wl, const float* __restrict__ Wg,
    const float* __restrict__ Wo, unsigned short* __restrict__ Wpre,
    unsigned int* __restrict__ tmax, unsigned int* __restrict__ outmax) {
  const int PER = HD * HD;  // 65536 elements per 256x256 matrix
  int t = blockIdx.x * 256 + threadIdx.x;
  if (t < 3 * NLAY * PER) {
    int mat = t / (NLAY * PER);
    int rem = t - mat * (NLAY * PER);
    int lay = rem / PER;
    int e   = rem & (PER - 1);
    int frag = e >> 9;
    int lane = (e >> 4) & 31;
    int j    = e & 15;
    int k = (frag & 7) * 32 + (j >> 3) * 16 + (lane >> 4) * 8 + (j & 7);   // B row (input channel)
    int n = (frag >> 3) * 16 + (lane & 15);           // B col (output channel)
    float s;
    if (mat == 0)      s = Wl[(size_t)lay * PER + (size_t)n * HD + k];
    else if (mat == 1) s = Wg[(size_t)lay * HD * 2 * HD + (size_t)n * 2 * HD + k];
    else               s = Wo[(size_t)n * (NLAY * HD) + (size_t)lay * HD + k];
    Wpre[t] = f2bf(s);
  }
  if (t < NLAY * BFB * HD) tmax[t] = 0u;   // post-ReLU values >= 0: identity is 0
  if (t < BFB * HD)        outmax[t] = 0u; // encf identity (below all finite encodings)
}

// ---------------- conv_in: [N,3] @ W_in^T, K=3 (VALU, trivial) ----------------
__global__ __launch_bounds__(256) void k_input(
    const float* __restrict__ pc, const float* __restrict__ W_in,
    const float* __restrict__ b_in, unsigned short* __restrict__ Y) {
  __shared__ float sw[HD * 3];
  __shared__ float sb[HD];
  for (int i = threadIdx.x; i < HD * 3; i += 256) sw[i] = W_in[i];
  for (int i = threadIdx.x; i < HD; i += 256) sb[i] = b_in[i];
  __syncthreads();
  const int p = blockIdx.x * 256 + threadIdx.x;
  const float x0 = pc[p * 3], x1 = pc[p * 3 + 1], x2 = pc[p * 3 + 2];
  unsigned short* yp = Y + (size_t)p * HD;
#pragma unroll 4
  for (int h = 0; h < HD; ++h) {
    float a = fmaf(x0, sw[h * 3], fmaf(x1, sw[h * 3 + 1], fmaf(x2, sw[h * 3 + 2], sb[h])));
    yp[h] = f2bf(fmaxf(a, 0.f));
  }
}

// ---------------- main bf16 WMMA GEMM + ReLU (+ optional channel max) --------
// Y = relu(X[131072,256] @ W^T + bias); wave tile 32x32, block 128 pts x 64 ch.
__global__ __launch_bounds__(256) void k_gemm_relu(
    const unsigned short* __restrict__ X, const unsigned short* __restrict__ Wfrag,
    const float* __restrict__ bias, int biasStride,
    unsigned short* __restrict__ Yo, unsigned int* __restrict__ tmax) {
  const int lane = threadIdx.x & 31, wave = threadIdx.x >> 5;
  const int row0 = blockIdx.x * 128 + (wave & 3) * 32;
  const int ch0  = blockIdx.y * 64 + (wave >> 2) * 32;
  const int bfi  = row0 >> 13;  // row / NPTS (NPTS=8192, blocks never straddle)
  const unsigned short* a0 = X + (size_t)row0 * HD;
  const unsigned short* a1 = a0 + (size_t)16 * HD;
  const int nt0 = ch0 >> 4;
  v8f c00 = {}, c01 = {}, c10 = {}, c11 = {};
#pragma unroll
  for (int kt = 0; kt < 8; ++kt) {
    v16bf af0 = load_a_frag(a0, kt * 32, lane);
    v16bf af1 = load_a_frag(a1, kt * 32, lane);
    v16bf bf0 = load_b_frag(Wfrag, nt0, kt, lane);
    v16bf bf1 = load_b_frag(Wfrag, nt0 + 1, kt, lane);
    c00 = WMMA_BF16(af0, bf0, c00);
    c01 = WMMA_BF16(af0, bf1, c01);
    c10 = WMMA_BF16(af1, bf0, c10);
    c11 = WMMA_BF16(af1, bf1, c11);
  }
  const int m0 = (lane >> 4) * 8;   // D layout: lanes<16 rows 0-7, lanes>=16 rows 8-15
  const int nlane = lane & 15;
#pragma unroll
  for (int t = 0; t < 2; ++t) {
    const int ch = ch0 + t * 16 + nlane;
    const float b = bias[biasStride * bfi + ch];
    v8f ct = t ? c01 : c00;
    v8f cb = t ? c11 : c10;
    float lmax = 0.f;
#pragma unroll
    for (int r = 0; r < 8; ++r) {
      float v = fmaxf(ct[r] + b, 0.f);
      Yo[(size_t)(row0 + m0 + r) * HD + ch] = f2bf(v);
      lmax = fmaxf(lmax, v);
      float w = fmaxf(cb[r] + b, 0.f);
      Yo[(size_t)(row0 + 16 + m0 + r) * HD + ch] = f2bf(w);
      lmax = fmaxf(lmax, w);
    }
    // post-ReLU values are non-negative: uint compare of float bits is exact
    if (tmax) atomicMax(&tmax[bfi * HD + ch], __float_as_uint(lmax));
  }
}

// ---------------- per-batch global-feature bias: g = tmax @ WgB^T + bg -------
__global__ __launch_bounds__(HD) void k_gvec(
    const unsigned int* __restrict__ tmax, const float* __restrict__ WgLayer,
    const float* __restrict__ bgLayer, float* __restrict__ gbias) {
  __shared__ float sx[HD];
  const int bfi = blockIdx.x;
  const int o = threadIdx.x;
  sx[o] = __uint_as_float(tmax[bfi * HD + o]);
  __syncthreads();
  float acc = bgLayer[o];
  const float* w = WgLayer + (size_t)o * (2 * HD) + HD;
#pragma unroll 8
  for (int h = 0; h < HD; ++h) acc = fmaf(sx[h], w[h], acc);
  gbias[bfi * HD + o] = acc;
}

// ---------------- conv_out accumulation GEMM (WMMA C-operand = fp32 accum) ---
// O += Y @ Wo_layer^T (layer 0 initializes with bias bo); layer 3 also does the
// masked per-(batch,channel) max via order-preserving atomicMax.
__global__ __launch_bounds__(256) void k_gemm_out(
    const unsigned short* __restrict__ X, const unsigned short* __restrict__ Wfrag,
    const float* __restrict__ bo, float* __restrict__ O, int layer,
    const unsigned char* __restrict__ mask, unsigned int* __restrict__ outmax) {
  const int lane = threadIdx.x & 31, wave = threadIdx.x >> 5;
  const int row0 = blockIdx.x * 128 + (wave & 3) * 32;
  const int ch0  = blockIdx.y * 64 + (wave >> 2) * 32;
  const int bfi  = row0 >> 13;
  const unsigned short* a0 = X + (size_t)row0 * HD;
  const unsigned short* a1 = a0 + (size_t)16 * HD;
  const int nt0 = ch0 >> 4;
  const int m0 = (lane >> 4) * 8;
  const int nlane = lane & 15;

  v8f c00, c01, c10, c11;
  if (layer == 0) {
    const float b0 = bo[ch0 + nlane], b1 = bo[ch0 + 16 + nlane];
#pragma unroll
    for (int r = 0; r < 8; ++r) { c00[r] = b0; c01[r] = b1; c10[r] = b0; c11[r] = b1; }
  } else {
#pragma unroll
    for (int r = 0; r < 8; ++r) {
      const size_t rt = (size_t)(row0 + m0 + r) * HD;
      const size_t rb = (size_t)(row0 + 16 + m0 + r) * HD;
      c00[r] = O[rt + ch0 + nlane];
      c01[r] = O[rt + ch0 + 16 + nlane];
      c10[r] = O[rb + ch0 + nlane];
      c11[r] = O[rb + ch0 + 16 + nlane];
    }
  }
#pragma unroll
  for (int kt = 0; kt < 8; ++kt) {
    v16bf af0 = load_a_frag(a0, kt * 32, lane);
    v16bf af1 = load_a_frag(a1, kt * 32, lane);
    v16bf bf0 = load_b_frag(Wfrag, nt0, kt, lane);
    v16bf bf1 = load_b_frag(Wfrag, nt0 + 1, kt, lane);
    c00 = WMMA_BF16(af0, bf0, c00);
    c01 = WMMA_BF16(af0, bf1, c01);
    c10 = WMMA_BF16(af1, bf0, c10);
    c11 = WMMA_BF16(af1, bf1, c11);
  }
#pragma unroll
  for (int t = 0; t < 2; ++t) {
    const int ch = ch0 + t * 16 + nlane;
    v8f ct = t ? c01 : c00;
    v8f cb = t ? c11 : c10;
    float mmax = -3.0e38f;
#pragma unroll
    for (int r = 0; r < 8; ++r) {
      const int rt = row0 + m0 + r, rb = rt + 16;
      O[(size_t)rt * HD + ch] = ct[r];
      O[(size_t)rb * HD + ch] = cb[r];
      if (layer == NLAY - 1) {
        float vt = mask[rt] ? ct[r] : ct[r] - 100000.f;
        float vb = mask[rb] ? cb[r] : cb[r] - 100000.f;
        mmax = fmaxf(mmax, fmaxf(vt, vb));
      }
    }
    if (layer == NLAY - 1) atomicMax(&outmax[bfi * HD + ch], encf(mmax));
  }
}

// ---------------- final projection: out = max_feat @ Wf^T + bf ----------------
__global__ __launch_bounds__(HD) void k_final(
    const unsigned int* __restrict__ outmax, const float* __restrict__ Wf,
    const float* __restrict__ bfv, float* __restrict__ out) {
  __shared__ float sx[HD];
  const int bfi = blockIdx.x;
  const int tid = threadIdx.x;
  sx[tid] = decf(outmax[bfi * HD + tid]);
  __syncthreads();
  const int o = blockIdx.y * HD + tid;
  float acc = bfv[o];
  const float* w = Wf + (size_t)o * HD;
#pragma unroll 8
  for (int h = 0; h < HD; ++h) acc = fmaf(sx[h], w[h], acc);
  out[(size_t)bfi * OUTC + o] = acc;
}

// ---------------- host-side orchestration ----------------
extern "C" void kernel_launch(void* const* d_in, const int* in_sizes, int n_in,
                              void* d_out, int out_size, void* d_ws, size_t ws_size,
                              hipStream_t stream) {
  // setup_inputs order: xyz, pc, mask, W_in, b_in, Wl, bl, Wg, bg, Wo, bo, Wf, bf
  const float* pc           = (const float*)d_in[1];
  const unsigned char* mask = (const unsigned char*)d_in[2];
  const float* W_in         = (const float*)d_in[3];
  const float* b_in         = (const float*)d_in[4];
  const float* Wl           = (const float*)d_in[5];
  const float* bl           = (const float*)d_in[6];
  const float* Wg           = (const float*)d_in[7];
  const float* bg           = (const float*)d_in[8];
  const float* Wo           = (const float*)d_in[9];
  const float* bo           = (const float*)d_in[10];
  const float* Wf           = (const float*)d_in[11];
  const float* bfv          = (const float*)d_in[12];
  float* out                = (float*)d_out;

  // workspace carve-out
  char* ws = (char*)d_ws;
  const size_t PER = (size_t)HD * HD;  // 65536
  unsigned short* Wpre = (unsigned short*)ws;                 // 3*NLAY*PER bf16
  size_t off = 3 * NLAY * PER * sizeof(unsigned short);
  unsigned int* tmax = (unsigned int*)(ws + off);  off += (size_t)NLAY * BFB * HD * 4;
  unsigned int* omax = (unsigned int*)(ws + off);  off += (size_t)BFB * HD * 4;
  float* gbias       = (float*)(ws + off);         off += (size_t)BFB * HD * 4;
  off = (off + 511) & ~(size_t)511;
  unsigned short* bufY = (unsigned short*)(ws + off); off += (size_t)TOTP * HD * 2; // 64MB
  unsigned short* bufT = (unsigned short*)(ws + off); off += (size_t)TOTP * HD * 2; // 64MB
  float* bufO          = (float*)(ws + off);          off += (size_t)TOTP * HD * 4; // 128MB

  unsigned short* WlB  = Wpre;
  unsigned short* WgAB = Wpre + NLAY * PER;
  unsigned short* WoB  = Wpre + 2 * NLAY * PER;

  k_init<<<dim3((unsigned)((3 * NLAY * PER + 255) / 256)), dim3(256), 0, stream>>>(
      Wl, Wg, Wo, Wpre, tmax, omax);
  k_input<<<dim3(TOTP / 256), dim3(256), 0, stream>>>(pc, W_in, b_in, bufY);

  const dim3 gg(TOTP / 128, HD / 64), bb(256);
  for (int i = 0; i < NLAY; ++i) {
    // t = relu(y @ Wl^T + bl), per-batch channel max -> tmax
    k_gemm_relu<<<gg, bb, 0, stream>>>(bufY, WlB + (size_t)i * PER, bl + i * HD, 0,
                                       bufT, tmax + i * BFB * HD);
    // fold y_global @ Wg[:,H:] + bg into per-batch bias
    k_gvec<<<dim3(BFB), dim3(HD), 0, stream>>>(tmax + i * BFB * HD,
                                               Wg + (size_t)i * HD * 2 * HD,
                                               bg + i * HD, gbias);
    // y = relu(t @ WgA^T + gbias)
    k_gemm_relu<<<gg, bb, 0, stream>>>(bufT, WgAB + (size_t)i * PER, gbias, HD,
                                       bufY, nullptr);
    // O (+)= y @ Wo_i^T ; last layer also does the masked global max
    k_gemm_out<<<gg, bb, 0, stream>>>(bufY, WoB + (size_t)i * PER, bo, bufO, i,
                                      mask, omax);
  }
  k_final<<<dim3(BFB, OUTC / HD), dim3(HD), 0, stream>>>(omax, Wf, bfv, out);
}
